// MixtureOfThoughtsLayer_74208444940560
// MI455X (gfx1250) — matrix operand order repaired and checked by reading, not compiled, not run on hardware
//
#include <hip/hip_runtime.h>
#include <hip/hip_bf16.h>

// ---------------- problem constants ----------------
#define B_    8
#define S_    512
#define H_    1024
#define T_    8
#define TOPK  2
#define FF_   4096
#define NH_   8
#define HD_   128
#define L_    (TOPK * S_)        // 1024
#define CREATIVITY_ 1.0f

typedef __bf16 bf16_t;
typedef __attribute__((ext_vector_type(8)))  bf16_t v8bf;
typedef __attribute__((ext_vector_type(16))) bf16_t v16bf;
typedef __attribute__((ext_vector_type(8)))  float  v8f;
typedef __attribute__((ext_vector_type(4)))  unsigned int v4u;
typedef __attribute__((ext_vector_type(4)))  int v4i;

// ---- CDNA5 async global->LDS path (ASYNCcnt-tracked DMA), with sync fallback ----
#if defined(__AMDGCN__) && defined(__has_builtin)
# if __has_builtin(__builtin_amdgcn_global_load_async_to_lds_b128)
#  define ASYNC_COPY 1
# endif
#endif
#ifndef ASYNC_COPY
# define ASYNC_COPY 0
#endif

#if ASYNC_COPY
static __device__ inline void async_ld16(const bf16_t* g, bf16_t* l) {
  // builtin signature: (v4i AS1* global_src, v4i AS3* lds_dst, imm offset, imm cpol)
  __builtin_amdgcn_global_load_async_to_lds_b128(
      (__attribute__((address_space(1))) v4i*)(void*)g,
      (__attribute__((address_space(3))) v4i*)(void*)l, 0, 0);
}
static __device__ inline void wait_async0() {
# if __has_builtin(__builtin_amdgcn_s_wait_asynccnt)
  __builtin_amdgcn_s_wait_asynccnt(0);
# else
  asm volatile("s_wait_asynccnt 0x0" ::: "memory");
# endif
}
#endif

// Build a 16-element bf16 fragment from two 16B chunks.
static __device__ inline v16bf load_frag2(const bf16_t* p0, const bf16_t* p1) {
  v8bf a = *reinterpret_cast<const v8bf*>(p0);
  v8bf b = *reinterpret_cast<const v8bf*>(p1);
  v16bf r;
#pragma unroll
  for (int i = 0; i < 8; ++i) { r[i] = a[i]; r[i + 8] = b[i]; }
  return r;
}

static __device__ inline v8f wmma_bf16(v16bf a, v16bf b, v8f c) {
  // (neg_a, A, neg_b, B, c_mod, C, reuse_a, reuse_b)
  return __builtin_amdgcn_wmma_f32_16x16x32_bf16(false, a, false, b, (short)0, c, false, false);
}

// ---------------- conversion / transpose ----------------
__global__ void conv_bf16_kernel(const float* __restrict__ src, bf16_t* __restrict__ dst, long long n) {
  long long i = (long long)blockIdx.x * blockDim.x + threadIdx.x;
  if (i < n) dst[i] = (bf16_t)src[i];
}

// dst[z][c][r] = (bf16) src[z][r][c]   (N-major "Bt" layout for GEMM B operands)
__global__ void transpose_conv_kernel(const float* __restrict__ src, bf16_t* __restrict__ dst,
                                      int R, int C) {
  __shared__ bf16_t tile[32][33];
  const long long zoff = (long long)blockIdx.z * R * C;
  src += zoff; dst += zoff;
  const int c0 = blockIdx.x * 32, r0 = blockIdx.y * 32;
  const int tx = threadIdx.x, ty = threadIdx.y;   // 32 x 8
#pragma unroll
  for (int i = 0; i < 4; ++i)
    tile[ty + i * 8][tx] = (bf16_t)src[(long long)(r0 + ty + i * 8) * C + c0 + tx];
  __syncthreads();
#pragma unroll
  for (int i = 0; i < 4; ++i)
    dst[(long long)(c0 + ty + i * 8) * R + r0 + tx] = tile[tx][ty + i * 8];
}

// ---------------- router ----------------
__global__ void pool_kernel(const float* __restrict__ hs, float* __restrict__ pooled) {
  int i = blockIdx.x * blockDim.x + threadIdx.x;   // b*H + h
  if (i >= B_ * H_) return;
  int b = i / H_, h = i % H_;
  const float* p = hs + (long long)b * S_ * H_ + h;
  float s = 0.f;
  for (int t = 0; t < S_; ++t) s += p[(long long)t * H_];
  pooled[i] = s * (1.0f / S_);
}

__global__ void router_kernel(const float* __restrict__ pooled,
                              const float* __restrict__ rw, const float* __restrict__ rb,
                              int* __restrict__ topi, float* __restrict__ wts,
                              float* __restrict__ ent_out) {
  __shared__ float pl[B_ * H_];
  __shared__ float logit[B_ * T_];
  __shared__ float prob[B_ * T_];
  __shared__ float ent[B_];
  const int tid = threadIdx.x;
  for (int i = tid; i < B_ * H_; i += blockDim.x) pl[i] = pooled[i];
  __syncthreads();
  if (tid < B_ * T_) {
    int b = tid / T_, t = tid % T_;
    float a = rb[t];
    for (int h = 0; h < H_; ++h) a += pl[b * H_ + h] * rw[h * T_ + t];
    logit[tid] = a;
  }
  __syncthreads();
  if (tid < B_) {
    int b = tid;
    float mx = -1e30f;
    for (int t = 0; t < T_; ++t) mx = fmaxf(mx, logit[b * T_ + t]);
    float s = 0.f;
    for (int t = 0; t < T_; ++t) { float e = expf(logit[b * T_ + t] - mx); prob[b * T_ + t] = e; s += e; }
    float inv = 1.f / s, en = 0.f;
    for (int t = 0; t < T_; ++t) {
      prob[b * T_ + t] *= inv;
      en -= prob[b * T_ + t] * logf(prob[b * T_ + t] + 1e-8f);
    }
    ent[b] = en;
    int i0 = 0;
    for (int t = 1; t < T_; ++t) if (prob[b * T_ + t] > prob[b * T_ + i0]) i0 = t;
    int i1 = (i0 == 0) ? 1 : 0;
    for (int t = 0; t < T_; ++t) if (t != i0 && prob[b * T_ + t] > prob[b * T_ + i1]) i1 = t;
    float p0 = prob[b * T_ + i0], p1 = prob[b * T_ + i1];
    float m2 = fmaxf(p0, p1);
    float e0 = expf(p0 - m2), e1 = expf(p1 - m2), s2 = e0 + e1;
    topi[b * TOPK + 0] = i0; topi[b * TOPK + 1] = i1;
    wts[b * TOPK + 0] = e0 / s2; wts[b * TOPK + 1] = e1 / s2;
  }
  __syncthreads();
  if (tid == 0) {
    float s = 0.f;
    for (int b = 0; b < B_; ++b) s += ent[b];
    ent_out[0] = s * (1.0f / B_);
  }
}

// ---------------- WMMA GEMM: C[z] = A[z] * Bt[zb]^T (+bias, gelu, scale) ----------------
// A: [M,Kd] bf16 row-major.  Bt: [N,Kd] bf16 (i.e. B transposed, N-major).
// All of M,N divisible by 128 and Kd by 32 here.
__global__ __launch_bounds__(256) void gemm_bf16_kernel(
    const bf16_t* __restrict__ A, long long strideA, int a_div,
    const bf16_t* __restrict__ Bt, long long strideB,
    const float* __restrict__ bias, long long strideBias,
    const int* __restrict__ emap,
    float* __restrict__ Cf, bf16_t* __restrict__ Cb, long long strideC,
    int M, int N, int Kd, float scale, int do_gelu)
{
#if ASYNC_COPY
  __shared__ bf16_t As[2][128][40];   // ping-pong, filled by async DMA
  __shared__ bf16_t Bs[2][128][40];
#else
  __shared__ bf16_t As[1][128][40];
  __shared__ bf16_t Bs[1][128][40];
#endif

  const int z  = blockIdx.z;
  const int zb = emap ? emap[z] : z;
  A  += (long long)(z / a_div) * strideA;
  Bt += (long long)zb * strideB;
  const float* biasp = bias ? bias + (long long)zb * strideBias : nullptr;

  const int tid  = threadIdx.x;
  const int lane = tid & 31;
  const int wv   = tid >> 5;
  const int wm   = wv >> 2;       // 0..1 : 64-row slab
  const int wn   = wv & 3;        // 0..3 : 32-col slab
  const int ln   = lane & 15;
  const int kh   = lane >> 4;

  const int m0 = blockIdx.y * 128;
  const int n0 = blockIdx.x * 128;

  const int srow = tid >> 1;           // 0..127
  const int sseg = (tid & 1) * 16;     // 0 | 16

  v8f acc[4][2];
#pragma unroll
  for (int i = 0; i < 4; ++i)
#pragma unroll
    for (int j = 0; j < 2; ++j)
#pragma unroll
      for (int r = 0; r < 8; ++r) acc[i][j][r] = 0.f;

  const int ksteps = Kd >> 5;

#if ASYNC_COPY
  auto stage = [&](int kt, int buf) {
    const bf16_t* srcA = A  + (long long)(m0 + srow) * Kd + (kt << 5) + sseg;
    const bf16_t* srcB = Bt + (long long)(n0 + srow) * Kd + (kt << 5) + sseg;
    async_ld16(srcA,     &As[buf][srow][sseg]);
    async_ld16(srcA + 8, &As[buf][srow][sseg + 8]);
    async_ld16(srcB,     &Bs[buf][srow][sseg]);
    async_ld16(srcB + 8, &Bs[buf][srow][sseg + 8]);
  };
  stage(0, 0);
#endif

  for (int kt = 0; kt < ksteps; ++kt) {
#if ASYNC_COPY
    wait_async0();            // our previous-stage DMAs have landed in LDS
    __syncthreads();          // everyone's have (dscnt flushed at barrier)
    if (kt + 1 < ksteps) stage(kt + 1, (kt + 1) & 1);  // overlap DMA with WMMA
    const int buf = kt & 1;
#else
    const int kb = kt << 5;
    __syncthreads();
    {
      const bf16_t* srcA = A + (long long)(m0 + srow) * Kd + kb + sseg;
      *reinterpret_cast<v4u*>(&As[0][srow][sseg])     = *reinterpret_cast<const v4u*>(srcA);
      *reinterpret_cast<v4u*>(&As[0][srow][sseg + 8]) = *reinterpret_cast<const v4u*>(srcA + 8);
      const bf16_t* srcB = Bt + (long long)(n0 + srow) * Kd + kb + sseg;
      *reinterpret_cast<v4u*>(&Bs[0][srow][sseg])     = *reinterpret_cast<const v4u*>(srcB);
      *reinterpret_cast<v4u*>(&Bs[0][srow][sseg + 8]) = *reinterpret_cast<const v4u*>(srcB + 8);
    }
    __syncthreads();
    const int buf = 0;
#endif

    v16bf af[4], bfm[2];
#pragma unroll
    for (int i = 0; i < 4; ++i) {
      // A-layout (16x32 bf16): lane m=ln, khalf kh: k in [kh*8 .. kh*8+7] and [16+kh*8 ..]
      const bf16_t* r0 = &As[buf][wm * 64 + i * 16 + ln][kh * 8];
      af[i] = load_frag2(r0, r0 + 16);
    }
#pragma unroll
    for (int j = 0; j < 2; ++j) {
      // B-layout (32x16 bf16): lane n=ln, khalf kh: k in [kh*8 .. kh*8+7] and [16+kh*8 ..]
      const bf16_t* r0 = &Bs[buf][wn * 32 + j * 16 + ln][kh * 8];
      bfm[j] = load_frag2(r0, r0 + 16);
    }
#pragma unroll
    for (int i = 0; i < 4; ++i)
#pragma unroll
      for (int j = 0; j < 2; ++j)
        acc[i][j] = wmma_bf16(af[i], bfm[j], acc[i][j]);
  }

  // epilogue: C layout VGPR r -> row (r + 8*kh), col ln within a 16x16 tile
#pragma unroll
  for (int j = 0; j < 2; ++j) {
    const int gn = n0 + wn * 32 + j * 16 + ln;
    const float bj = biasp ? biasp[gn] : 0.f;
#pragma unroll
    for (int i = 0; i < 4; ++i) {
      const int gm_base = m0 + wm * 64 + i * 16 + kh * 8;
#pragma unroll
      for (int r = 0; r < 8; ++r) {
        float v = acc[i][j][r] + bj;
        if (do_gelu) v = 0.5f * v * (1.0f + erff(v * 0.70710678118654752f));
        v *= scale;
        const long long idx = (long long)z * strideC + (long long)(gm_base + r) * N + gn;
        if (Cf) Cf[idx] = v;
        if (Cb) Cb[idx] = (bf16_t)v;
      }
    }
  }
}

// ---------------- flash attention combiner ----------------
// q/k/v/o: bf16 [B][L][H], head h occupies columns [h*HD, (h+1)*HD). 1/sqrt(HD) pre-folded into q.
__global__ __launch_bounds__(256) void attn_kernel(
    const bf16_t* __restrict__ Q, const bf16_t* __restrict__ Km,
    const bf16_t* __restrict__ V, bf16_t* __restrict__ O)
{
  __shared__ bf16_t Vt[128][40];      // V^T for the current 32-key block: [d][key]
  __shared__ bf16_t Pb[8][16][40];    // per-wave P tile (16 x 32)
#if ASYNC_COPY
  __shared__ bf16_t Ks[32][136];      // raw K rows for current block: [key][d]
#endif

  const int b    = blockIdx.z;
  const int head = blockIdx.y;
  const int tid  = threadIdx.x;
  const int lane = tid & 31;
  const int wv   = tid >> 5;
  const int ln   = lane & 15;
  const int kh   = lane >> 4;

  const bf16_t* Qp = Q + (long long)b * L_ * H_ + head * HD_;
  const bf16_t* Kp = Km + (long long)b * L_ * H_ + head * HD_;
  const bf16_t* Vp = V + (long long)b * L_ * H_ + head * HD_;
  bf16_t*       Op = O + (long long)b * L_ * H_ + head * HD_;

  const int qrow0 = blockIdx.x * 128 + wv * 16;

  // resident Q fragments (16 rows x HD, 4 K-steps of 32)
  v16bf qf[4];
#pragma unroll
  for (int j = 0; j < 4; ++j) {
    const bf16_t* base = Qp + (long long)(qrow0 + ln) * H_ + j * 32 + kh * 8;
    qf[j] = load_frag2(base, base + 16);
  }

  v8f of[8];
  float mr[8], lr[8];
#pragma unroll
  for (int t = 0; t < 8; ++t)
#pragma unroll
    for (int r = 0; r < 8; ++r) of[t][r] = 0.f;
#pragma unroll
  for (int r = 0; r < 8; ++r) { mr[r] = -1e30f; lr[r] = 0.f; }

  const int vkey = tid >> 3;           // 0..31
  const int vd0  = (tid & 7) * 16;     // 0..112

  for (int it = 0; it < L_ / 32; ++it) {
    const int kb0 = it * 32;
    __syncthreads();
    {   // stage V^T (32 keys x 128 dims, transposed)
      const bf16_t* src = Vp + (long long)(kb0 + vkey) * H_ + vd0;
      v8bf x0 = *reinterpret_cast<const v8bf*>(src);
      v8bf x1 = *reinterpret_cast<const v8bf*>(src + 8);
#pragma unroll
      for (int e = 0; e < 8; ++e) Vt[vd0 + e][vkey] = x0[e];
#pragma unroll
      for (int e = 0; e < 8; ++e) Vt[vd0 + 8 + e][vkey] = x1[e];
    }
#if ASYNC_COPY
    {   // async-DMA the 32x128 K block straight into LDS (shared by all 8 waves)
      const bf16_t* src = Kp + (long long)(kb0 + vkey) * H_ + vd0;
      async_ld16(src,     &Ks[vkey][vd0]);
      async_ld16(src + 8, &Ks[vkey][vd0 + 8]);
    }
    wait_async0();
#endif
    __syncthreads();

    // S = Q * K^T  (two 16-key tiles)
    v8f s[2];
#pragma unroll
    for (int nt = 0; nt < 2; ++nt) {
#pragma unroll
      for (int r = 0; r < 8; ++r) s[nt][r] = 0.f;
#pragma unroll
      for (int j = 0; j < 4; ++j) {
#if ASYNC_COPY
        const bf16_t* base = &Ks[nt * 16 + ln][j * 32 + kh * 8];
#else
        const bf16_t* base = Kp + (long long)(kb0 + nt * 16 + ln) * H_ + j * 32 + kh * 8;
#endif
        v16bf kf = load_frag2(base, base + 16);
        s[nt] = wmma_bf16(qf[j], kf, s[nt]);
      }
    }

    // online softmax; row of VGPR r is (r + 8*kh); reductions stay in 16-lane halves
#pragma unroll
    for (int r = 0; r < 8; ++r) {
      float x0 = s[0][r], x1 = s[1][r];
      float mx = fmaxf(x0, x1);
#pragma unroll
      for (int d = 8; d >= 1; d >>= 1) mx = fmaxf(mx, __shfl_xor(mx, d, 32));
      float mnew  = fmaxf(mr[r], mx);
      float alpha = __expf(mr[r] - mnew);
      float p0 = __expf(x0 - mnew);
      float p1 = __expf(x1 - mnew);
      float rs = p0 + p1;
#pragma unroll
      for (int d = 8; d >= 1; d >>= 1) rs += __shfl_xor(rs, d, 32);
      lr[r] = lr[r] * alpha + rs;
      mr[r] = mnew;
#pragma unroll
      for (int t = 0; t < 8; ++t) of[t][r] *= alpha;
      s[0][r] = p0; s[1][r] = p1;
    }

    // bounce P through wave-private LDS to convert C-layout -> A-layout (in-order per wave)
#pragma unroll
    for (int nt = 0; nt < 2; ++nt)
#pragma unroll
      for (int r = 0; r < 8; ++r)
        Pb[wv][r + 8 * kh][nt * 16 + ln] = (bf16_t)s[nt][r];

    const bf16_t* pr0 = &Pb[wv][ln][kh * 8];
    v16bf pf = load_frag2(pr0, pr0 + 16);

    // O += P * V
#pragma unroll
    for (int t = 0; t < 8; ++t) {
      const bf16_t* vb0 = &Vt[t * 16 + ln][kh * 8];
      v16bf vf = load_frag2(vb0, vb0 + 16);
      of[t] = wmma_bf16(pf, vf, of[t]);
    }
  }

  // epilogue: normalize and store ctx
#pragma unroll
  for (int t = 0; t < 8; ++t)
#pragma unroll
    for (int r = 0; r < 8; ++r) {
      const float v = of[t][r] / lr[r];
      const int row = qrow0 + r + 8 * kh;
      Op[(long long)row * H_ + t * 16 + ln] = (bf16_t)v;
    }
}

// ---------------- combine + metrics ----------------
__global__ void combine_kernel(const float* __restrict__ att, const float* __restrict__ wts,
                               float* __restrict__ out) {
  long long i = (long long)blockIdx.x * blockDim.x + threadIdx.x;
  if (i >= (long long)B_ * S_ * H_) return;
  int b = (int)(i / ((long long)S_ * H_));
  long long r = i % ((long long)S_ * H_);
  long long base = (long long)b * TOPK * S_ * H_;
  out[i] = wts[b * TOPK + 0] * att[base + r] +
           wts[b * TOPK + 1] * att[base + (long long)S_ * H_ + r];
}

__global__ void diversity_partial_kernel(const float* __restrict__ branchf, float* __restrict__ acc3) {
  __shared__ float sd[256], s0[256], s1[256];
  const int b = blockIdx.y;
  const float* a = branchf + (long long)b * TOPK * S_ * H_;
  const float* c = a + (long long)S_ * H_;
  long long base = (long long)blockIdx.x * 2048 + threadIdx.x;
  float d = 0.f, n0 = 0.f, n1 = 0.f;
#pragma unroll
  for (int i = 0; i < 8; ++i) {
    long long idx = base + (long long)i * 256;
    float x = a[idx], y = c[idx];
    d += x * y; n0 += x * x; n1 += y * y;
  }
  sd[threadIdx.x] = d; s0[threadIdx.x] = n0; s1[threadIdx.x] = n1;
  __syncthreads();
  for (int st = 128; st > 0; st >>= 1) {
    if ((int)threadIdx.x < st) {
      sd[threadIdx.x] += sd[threadIdx.x + st];
      s0[threadIdx.x] += s0[threadIdx.x + st];
      s1[threadIdx.x] += s1[threadIdx.x + st];
    }
    __syncthreads();
  }
  if (threadIdx.x == 0) {
    atomicAdd(&acc3[b * 3 + 0], sd[0]);
    atomicAdd(&acc3[b * 3 + 1], s0[0]);
    atomicAdd(&acc3[b * 3 + 2], s1[0]);
  }
}

__global__ void diversity_final_kernel(const float* __restrict__ acc3, float* __restrict__ out) {
  if (threadIdx.x == 0 && blockIdx.x == 0) {
    float s = 0.f;
    for (int b = 0; b < B_; ++b) {
      float dot = acc3[b * 3 + 0];
      float nn  = sqrtf(acc3[b * 3 + 1]) * sqrtf(acc3[b * 3 + 2]);
      s += 1.f - dot / fmaxf(nn, 1e-8f);
    }
    out[0] = s * (1.0f / B_);
  }
}

// ---------------- host orchestration ----------------
extern "C" void kernel_launch(void* const* d_in, const int* in_sizes, int n_in,
                              void* d_out, int out_size, void* d_ws, size_t ws_size,
                              hipStream_t stream) {
  (void)in_sizes; (void)n_in; (void)out_size; (void)ws_size;
  const float* hs = (const float*)d_in[0];
  const float* rw = (const float*)d_in[1];
  const float* rb = (const float*)d_in[2];
  const float* W1 = (const float*)d_in[3];
  const float* b1 = (const float*)d_in[4];
  const float* W2 = (const float*)d_in[5];
  const float* b2 = (const float*)d_in[6];
  const float* Wq = (const float*)d_in[7];
  const float* bq = (const float*)d_in[8];
  const float* Wk = (const float*)d_in[9];
  const float* bk = (const float*)d_in[10];
  const float* Wv = (const float*)d_in[11];
  const float* bv = (const float*)d_in[12];
  const float* Wo = (const float*)d_in[13];
  const float* bo = (const float*)d_in[14];
  float* outp = (float*)d_out;

  char* ws = (char*)d_ws;
  size_t off = 0;
  auto alloc = [&](size_t bytes) { size_t o = off; off += (bytes + 255) & ~(size_t)255; return o; };

  bf16_t* hsb  = (bf16_t*)(ws + alloc((size_t)B_ * S_ * H_ * 2));
  bf16_t* W1b  = (bf16_t*)(ws + alloc((size_t)T_ * H_ * FF_ * 2));   // [T][FF][H]
  bf16_t* W2b  = (bf16_t*)(ws + alloc((size_t)T_ * FF_ * H_ * 2));   // [T][H][FF]
  bf16_t* Wqb  = (bf16_t*)(ws + alloc((size_t)H_ * H_ * 2));
  bf16_t* Wkb  = (bf16_t*)(ws + alloc((size_t)H_ * H_ * 2));
  bf16_t* Wvb  = (bf16_t*)(ws + alloc((size_t)H_ * H_ * 2));
  bf16_t* Wob  = (bf16_t*)(ws + alloc((size_t)H_ * H_ * 2));
  bf16_t* actb = (bf16_t*)(ws + alloc((size_t)B_ * TOPK * S_ * FF_ * 2));
  float*  brf  = (float*) (ws + alloc((size_t)B_ * TOPK * S_ * H_ * 4));
  bf16_t* brb  = (bf16_t*)(ws + alloc((size_t)B_ * TOPK * S_ * H_ * 2));
  float*  pooled = (float*)(ws + alloc((size_t)B_ * H_ * 4));
  int*    topi   = (int*)  (ws + alloc((size_t)B_ * TOPK * 4));
  float*  wts    = (float*)(ws + alloc((size_t)B_ * TOPK * 4));
  float*  dacc   = (float*)(ws + alloc((size_t)B_ * 3 * 4));
  // Safe overlays (each region is re-produced earlier in every call than it is consumed):
  const size_t qkvBytes = (size_t)B_ * L_ * H_ * 2;  // 16 MB each; W1b region = 64 MB
  bf16_t* qb   = (bf16_t*)((char*)W1b);
  bf16_t* kb   = (bf16_t*)((char*)W1b + qkvBytes);
  bf16_t* vb   = (bf16_t*)((char*)W1b + 2 * qkvBytes);
  bf16_t* ctxb = (bf16_t*)((char*)actb);
  float*  attf = (float*) ((char*)actb + qkvBytes);

  // 1) conversions (every call; overlays rely on this ordering)
  {
    long long n = (long long)B_ * S_ * H_;
    conv_bf16_kernel<<<dim3((unsigned)((n + 255) / 256)), 256, 0, stream>>>(hs, hsb, n);
  }
  transpose_conv_kernel<<<dim3(FF_ / 32, H_ / 32, T_), dim3(32, 8), 0, stream>>>(W1, W1b, H_, FF_);
  transpose_conv_kernel<<<dim3(H_ / 32, FF_ / 32, T_), dim3(32, 8), 0, stream>>>(W2, W2b, FF_, H_);
  transpose_conv_kernel<<<dim3(H_ / 32, H_ / 32, 1), dim3(32, 8), 0, stream>>>(Wq, Wqb, H_, H_);
  transpose_conv_kernel<<<dim3(H_ / 32, H_ / 32, 1), dim3(32, 8), 0, stream>>>(Wk, Wkb, H_, H_);
  transpose_conv_kernel<<<dim3(H_ / 32, H_ / 32, 1), dim3(32, 8), 0, stream>>>(Wv, Wvb, H_, H_);
  transpose_conv_kernel<<<dim3(H_ / 32, H_ / 32, 1), dim3(32, 8), 0, stream>>>(Wo, Wob, H_, H_);

  // 2) router (device-side top-k; writes entropy straight into d_out)
  pool_kernel<<<dim3((B_ * H_ + 255) / 256), 256, 0, stream>>>(hs, pooled);
  router_kernel<<<1, 256, 0, stream>>>(pooled, rw, rb, topi, wts, outp + (long long)B_ * S_ * H_);

  // 3) expert FFN: h = gelu(hs @ W1[e] + b1[e]);  branch = (h @ W2[e] + b2[e]) * CREATIVITY
  gemm_bf16_kernel<<<dim3(FF_ / 128, S_ / 128, B_ * TOPK), 256, 0, stream>>>(
      hsb, (long long)S_ * H_, TOPK,
      W1b, (long long)FF_ * H_, b1, (long long)FF_, topi,
      nullptr, actb, (long long)S_ * FF_,
      S_, FF_, H_, 1.0f, 1);
  gemm_bf16_kernel<<<dim3(H_ / 128, S_ / 128, B_ * TOPK), 256, 0, stream>>>(
      actb, (long long)S_ * FF_, 1,
      W2b, (long long)H_ * FF_, b2, (long long)H_, topi,
      brf, brb, (long long)S_ * H_,
      S_, H_, FF_, CREATIVITY_, 0);

  // 4) Q/K/V projections over flat = branch.reshape(B, L, H); 1/sqrt(HD) folded into q
  const float qscale = 0.08838834764831845f;  // 1/sqrt(128)
  gemm_bf16_kernel<<<dim3(H_ / 128, L_ / 128, B_), 256, 0, stream>>>(
      brb, (long long)L_ * H_, 1, Wqb, 0, bq, 0, nullptr,
      nullptr, qb, (long long)L_ * H_, L_, H_, H_, qscale, 0);
  gemm_bf16_kernel<<<dim3(H_ / 128, L_ / 128, B_), 256, 0, stream>>>(
      brb, (long long)L_ * H_, 1, Wkb, 0, bk, 0, nullptr,
      nullptr, kb, (long long)L_ * H_, L_, H_, H_, 1.0f, 0);
  gemm_bf16_kernel<<<dim3(H_ / 128, L_ / 128, B_), 256, 0, stream>>>(
      brb, (long long)L_ * H_, 1, Wvb, 0, bv, 0, nullptr,
      nullptr, vb, (long long)L_ * H_, L_, H_, H_, 1.0f, 0);

  // 5) flash attention
  attn_kernel<<<dim3(L_ / 128, NH_, B_), 256, 0, stream>>>(qb, kb, vb, ctxb);

  // 6) output projection: attended = ctx @ Wo + bo
  gemm_bf16_kernel<<<dim3(H_ / 128, L_ / 128, B_), 256, 0, stream>>>(
      ctxb, (long long)L_ * H_, 1, Wob, 0, bo, 0, nullptr,
      attf, nullptr, (long long)L_ * H_, L_, H_, H_, 1.0f, 0);

  // 7) weighted combine over k -> d_out[0 : B*S*H]
  combine_kernel<<<dim3((unsigned)(((long long)B_ * S_ * H_ + 255) / 256)), 256, 0, stream>>>(
      attf, wts, outp);

  // 8) diversity metric
  (void)hipMemsetAsync(dacc, 0, (size_t)B_ * 3 * sizeof(float), stream);
  diversity_partial_kernel<<<dim3(S_ * H_ / 2048, B_), 256, 0, stream>>>(brf, dacc);
  diversity_final_kernel<<<1, 32, 0, stream>>>(dacc, outp + (long long)B_ * S_ * H_ + 1);
}
